// NeighborSelectionPPORLlib_30382598652484
// MI455X (gfx1250) — matrix operand order repaired and checked by reading, not compiled, not run on hardware
//
#include <hip/hip_runtime.h>

// ============================================================================
// NeighborSelectionPPORLlib forward for MI455X (gfx1250, wave32, WMMA).
//
// Shapes: B=32, N=64, OBS=32, D=128, DFF=512, HEADS=8, DH=16.
// 2048 = N*B independent sequences of 64 tokens.
//
// Precision: all GEMMs via v_wmma_f32_16x16x32_f16 (f16 in, f32 accum);
// softmax / layernorm / reductions in f32.
//
// Weights are converted once (prep kernel) to f16 AND pre-swizzled into the
// WMMA B-fragment VGPR order: per (32x16) K-N tile, lane l's 16 halves are
// contiguous, so a B fragment is one aligned 32-byte vector load per lane
// (2x global_load_b128) instead of 16 strided global_load_u16.
//
// d_in layout assumption: top-level arrays in setup_inputs() insertion order
// (agent_infos, neighbor_masks, padding_mask, is_from_my_env), with a runtime
// fallback (via in_sizes) for jax-sorted order. 'params' leaves follow in JAX
// pytree (sorted-key) order:
//   4:We 5:be | dec: 6:W1 7:W2 8:Wk 9:Wo 10:Wq 11:Wv 12:b1 13:b2 14:bb1 15:bb2
//   16:g1 17:g2 | enc: 18:W1 19:W2 20:Wk 21:Wo 22:Wq 23:Wv 24:b1 25:b2 26:bb1
//   27:bb2 28:g1 29:g2 | gen: 30:Wk 31:Wq
//
// Workspace usage: ~35 MB (f16 weights 860 KB, h_c 1 MB, attn_out 1 MB,
// gen-keys f16 32 MB).
// ============================================================================

typedef _Float16 half_t;
typedef __attribute__((ext_vector_type(16))) _Float16 v16h;
typedef __attribute__((ext_vector_type(8)))  float    v8f;

#define NEGBIG (-1.0e9f)

__device__ __forceinline__ v8f wmma16(v16h a, v16h b, v8f c) {
  // D = A(16x32 f16) * B(32x16 f16) + C(16x16 f32)
  return __builtin_amdgcn_wmma_f32_16x16x32_f16(false, a, false, b, (short)0, c,
                                                false, false);
}

// --- fragment builders (A/B: contiguous k = 16*(lane/16)+h, wave32) --------------
// A (16x32 f16): lane -> M = lane&15; half h -> K = (lane&16?16:0)+h
__device__ __forceinline__ v16h frag_a_f32(const float* src, int ld, int row0,
                                           int k0, int klim) {
  int lane = threadIdx.x & 31;
  int m = lane & 15;
  int kb = (lane & 16) ? 16 : 0;
  v16h a;
#pragma unroll
  for (int h = 0; h < 16; ++h) {
    int kk = kb + h;
    a[h] = (half_t)((kk < klim) ? src[(row0 + m) * ld + k0 + kk] : 0.0f);
  }
  return a;
}

// B fragment from pre-swizzled f16 weights: one 32-byte vector load per lane.
// Swizzled layout: tile (kt,nt) of a KxN weight -> 32 lanes x 16 halves,
// lane = (n&15) + (k&16), h = k&15  (K = kt*32+k_local, N = nt*16+n_local).
__device__ __forceinline__ v16h frag_b_sw(const half_t* base, int ntiles,
                                          int kt, int nt) {
  int lane = threadIdx.x & 31;
  const v16h* p = (const v16h*)base + ((kt * ntiles + nt) * 32 + lane);
  return *p;
}

// B (32x16) from f32 LDS row-major (for P @ V)
__device__ __forceinline__ v16h frag_b_f32(const float* src, int ld, int k0,
                                           int col0, int klim) {
  int lane = threadIdx.x & 31;
  int nn = lane & 15;
  int kb = (lane & 16) ? 16 : 0;
  v16h b;
#pragma unroll
  for (int h = 0; h < 16; ++h) {
    int kk = kb + h;
    b[h] = (half_t)((kk < klim) ? src[(k0 + kk) * ld + col0 + nn] : 0.0f);
  }
  return b;
}

// B = M^T : B[k][n] = src[(row0+n)*ld + c0 + k]   (for Q*K^T)
__device__ __forceinline__ v16h frag_bT_f32(const float* src, int ld, int row0,
                                            int c0, int klim) {
  int lane = threadIdx.x & 31;
  int nn = lane & 15;
  int kb = (lane & 16) ? 16 : 0;
  v16h b;
#pragma unroll
  for (int h = 0; h < 16; ++h) {
    int kk = kb + h;
    b[h] = (half_t)((kk < klim) ? src[(row0 + nn) * ld + c0 + kk] : 0.0f);
  }
  return b;
}

// C/D (16x16 f32): VGPR v -> M = (lane&16?8:0)+v, N = lane&15
__device__ __forceinline__ void store_c(float* dst, int ld, int row0, int col0,
                                        v8f c, const float* bias, bool relu,
                                        float scale) {
  int lane = threadIdx.x & 31;
  int nn = lane & 15;
  int rb = (lane & 16) ? 8 : 0;
  float bv = bias ? bias[col0 + nn] : 0.0f;
#pragma unroll
  for (int v = 0; v < 8; ++v) {
    float x = c[v] * scale + bv;
    if (relu) x = fmaxf(x, 0.0f);
    dst[(row0 + rb + v) * ld + col0 + nn] = x;
  }
}

// C(MxN f32 LDS) = A(MxK f32 LDS) * Bw(swizzled f16 global) [+bias][relu]
// Bw tile (kt0 + k0/32, nt0 + tn); ntiles = full weight N / 16.
// 8 waves; each wave owns tiles t = w, w+8, ...   K multiple of 32.
__device__ __forceinline__ void gemm_sw(const float* A, int lda,
                                        const half_t* Bw, int ntiles, int nt0,
                                        int kt0, float* C, int ldc, int M,
                                        int N, int K, const float* bias,
                                        bool relu) {
  int w = (int)(threadIdx.x >> 5);
  int mt = M >> 4, ntl = N >> 4;
  for (int t = w; t < mt * ntl; t += 8) {
    int tm = t / ntl, tn = t % ntl;
    v8f c = {};
    for (int k0 = 0; k0 < K; k0 += 32) {
      v16h a = frag_a_f32(A, lda, tm * 16, k0, 32);
      v16h b = frag_b_sw(Bw, ntiles, kt0 + (k0 >> 5), nt0 + tn);
      c = wmma16(a, b, c);
    }
    store_c(C, ldc, tm * 16, tn * 16, c, bias, relu, 1.0f);
  }
}

// LayerNorm of (x+r) over 64 rows x 128 cols; one row per thread (tid<64).
__device__ __forceinline__ void ln_rows64(const float* x, const float* r,
                                          float* o, const float* g,
                                          const float* b) {
  int tid = threadIdx.x;
  if (tid < 64) {
    const float* xr = x + tid * 128;
    const float* rr = r + tid * 128;
    float* orow = o + tid * 128;
    float s = 0.f, s2 = 0.f;
    for (int j = 0; j < 128; ++j) {
      float v = xr[j] + rr[j];
      s += v;
      s2 += v * v;
    }
    float m = s * (1.0f / 128.0f);
    float var = fmaxf(s2 * (1.0f / 128.0f) - m * m, 0.0f);
    float rs = rsqrtf(var + 1e-5f);
    for (int j = 0; j < 128; ++j) {
      float v = xr[j] + rr[j];
      orow[j] = (v - m) * rs * g[j] + b[j];
    }
  }
}

// ================= prep: weights -> f16, swizzled to B-fragment order =======
// Table 9 (dec Wq) is instead stored TRANSPOSED (row-major 128x128, out-major)
// for the VALU decoder-query dot.
struct PrepArgs { const float* src[15]; };

__global__ void prep_kernel(PrepArgs pa, half_t* w16) {
  const int cntK[15] = {32, 128, 128, 128, 128, 128, 512, 128,
                        128, 128, 128, 128, 512, 128, 128};
  const int cntN[15] = {128, 128, 128, 128, 128, 512, 128, 128,
                        128, 128, 128, 512, 128, 128, 128};
  int gid = blockIdx.x * 256 + threadIdx.x;
  int stride = gridDim.x * 256;
  int off = 0;
#pragma unroll
  for (int t = 0; t < 15; ++t) {
    const int K = cntK[t], N = cntN[t];
    const float* s = pa.src[t];
    for (int i = gid; i < K * N; i += stride) {
      int k = i / N, n = i - k * N;
      int di;
      if (t == 9) {
        di = n * K + k;  // transpose
      } else {
        int kt = k >> 5, kk = k & 31, nt = n >> 4, nn = n & 15;
        di = (((kt * (N >> 4) + nt) * 32) + nn + (kk & 16)) * 16 + (kk & 15);
      }
      w16[off + di] = (half_t)s[i];
    }
    off += K * N;
  }
}

// ======================= kernel A: per-sequence encoder =====================
struct AArgs {
  const float* agent;
  const int* nmask;
  const half_t* WE;
  const float* be;
  const half_t *EWQ, *EWK, *EWV, *EWO, *EW1, *EW2;
  const float *eb1, *eb2, *eg1, *ebb1, *eg2, *ebb2;
  const half_t *DWK, *DWV, *DWQT, *GWK;
  float* hcws;
  float* aows;
  half_t* ekws;
};

__global__ __launch_bounds__(256) void enc_kernel(AArgs A) {
  __shared__ float sA[64 * 128];   // h / h1
  __shared__ float sBq[64 * 128];  // Q -> O -> dec K -> gen keys
  __shared__ float sCk[64 * 128];  // K -> attn proj -> FFN blk -> dec V
  __shared__ float sDv[64 * 128];  // V -> enc
  __shared__ float sS[64 * 64];    // x input / attention probs / dec probs
  __shared__ float sTok[64];
  __shared__ float sVec[512];  // [0:128) h_c  [128:256) q_d  [256:384) attn_out
  __shared__ float sScal[2];

  int tid = threadIdx.x;
  int fb = blockIdx.x;  // fb = n*32 + b
  int nag = fb >> 5;
  int bb = fb & 31;
  int w = tid >> 5;

  // 1. load x (64x32) + token mask
  const float* xin = A.agent + (size_t)(bb * 64 + nag) * 2048;
  for (int i = tid; i < 64 * 32; i += 256) sS[i] = xin[i];
  if (tid < 64)
    sTok[tid] = (A.nmask[(bb * 64 + nag) * 64 + tid] != 0) ? 1.0f : 0.0f;
  __syncthreads();
  if (tid == 0) {
    float c = 0;
    for (int k = 0; k < 64; ++k) c += sTok[k];
    sScal[0] = fmaxf(c, 1.0f);
  }
  // 2. embed: h = x @ We + be
  gemm_sw(sS, 32, A.WE, 8, 0, 0, sA, 128, 64, 128, 32, A.be, false);
  __syncthreads();
  // 3. Q, K, V
  gemm_sw(sA, 128, A.EWQ, 8, 0, 0, sBq, 128, 64, 128, 128, nullptr, false);
  gemm_sw(sA, 128, A.EWK, 8, 0, 0, sCk, 128, 64, 128, 128, nullptr, false);
  gemm_sw(sA, 128, A.EWV, 8, 0, 0, sDv, 128, 64, 128, 128, nullptr, false);
  __syncthreads();
  // 4. attention heads (dh=16, K padded to 32 with zeros)
  for (int hd = 0; hd < 8; ++hd) {
    for (int t = w; t < 16; t += 8) {  // 4x4 tiles of 64x64 logits
      int tm = t >> 2, tn = t & 3;
      v16h a = frag_a_f32(sBq, 128, tm * 16, hd * 16, 16);
      v16h b = frag_bT_f32(sCk, 128, tn * 16, hd * 16, 16);
      v8f c = {};
      c = wmma16(a, b, c);
      store_c(sS, 64, tm * 16, tn * 16, c, nullptr, false, 0.25f);  // /sqrt(16)
    }
    __syncthreads();
    if (tid < 64) {  // masked softmax, one row per thread
      float tq = sTok[tid];
      float mx = -3.4e38f;
      for (int k = 0; k < 64; ++k) {
        float v = (tq > 0.5f && sTok[k] > 0.5f) ? sS[tid * 64 + k] : NEGBIG;
        mx = fmaxf(mx, v);
      }
      float su = 0.f;
      for (int k = 0; k < 64; ++k) {
        float v = (tq > 0.5f && sTok[k] > 0.5f) ? sS[tid * 64 + k] : NEGBIG;
        float e = __expf(v - mx);
        sS[tid * 64 + k] = e;
        su += e;
      }
      float inv = 1.0f / su;
      for (int k = 0; k < 64; ++k) sS[tid * 64 + k] *= inv;
    }
    __syncthreads();
    if (w < 4) {  // O[:,hd] = P @ V[:,hd]  (writes into dead Q head columns)
      v8f c = {};
      for (int ks = 0; ks < 2; ++ks) {
        v16h a = frag_a_f32(sS, 64, w * 16, ks * 32, 32);
        v16h b = frag_b_f32(sDv, 128, ks * 32, hd * 16, 32);
        c = wmma16(a, b, c);
      }
      store_c(sBq, 128, w * 16, hd * 16, c, nullptr, false, 1.0f);
    }
    __syncthreads();
  }
  // 5. attn projection: T = O @ Wo
  gemm_sw(sBq, 128, A.EWO, 8, 0, 0, sCk, 128, 64, 128, 128, nullptr, false);
  __syncthreads();
  // 6. h1 = LN(h + T)
  ln_rows64(sA, sCk, sA, A.eg1, A.ebb1);
  __syncthreads();
  // 7. FFN: out = relu(h1@W1+b1)@W2 + b2  (blocked over K=512, reg accum)
  v8f acc[4];
  {
    v8f z = {};
#pragma unroll
    for (int i = 0; i < 4; ++i) acc[i] = z;
  }
  for (int kb = 0; kb < 4; ++kb) {
    gemm_sw(sA, 128, A.EW1, 32, kb * 8, 0, sCk, 128, 64, 128, 128,
            A.eb1 + kb * 128, true);
    __syncthreads();
#pragma unroll
    for (int ti = 0; ti < 4; ++ti) {
      int t = w + ti * 8;
      int tm = t >> 3, tn = t & 7;
      for (int ks = 0; ks < 4; ++ks) {
        v16h a = frag_a_f32(sCk, 128, tm * 16, ks * 32, 32);
        v16h b = frag_b_sw(A.EW2, 8, kb * 4 + ks, tn);
        acc[ti] = wmma16(a, b, acc[ti]);
      }
    }
    __syncthreads();
  }
#pragma unroll
  for (int ti = 0; ti < 4; ++ti) {
    int t = w + ti * 8;
    store_c(sDv, 128, (t >> 3) * 16, (t & 7) * 16, acc[ti], A.eb2, false, 1.0f);
  }
  __syncthreads();
  // 8. enc = LN(h1 + ffn)
  ln_rows64(sA, sDv, sDv, A.eg2, A.ebb2);
  __syncthreads();
  // 9. h_c = sum(enc * tok) / cnt
  if (tid < 128) {
    float s = 0;
    for (int i = 0; i < 64; ++i) s += sDv[i * 128 + tid] * sTok[i];
    sVec[tid] = s / sScal[0];
  }
  __syncthreads();
  // 10. decoder K, V
  gemm_sw(sDv, 128, A.DWK, 8, 0, 0, sBq, 128, 64, 128, 128, nullptr, false);
  gemm_sw(sDv, 128, A.DWV, 8, 0, 0, sCk, 128, 64, 128, 128, nullptr, false);
  __syncthreads();
  // 11. decoder attention (Lq = 1); DWQT is transposed -> contiguous reads
  if (tid < 128) {
    const v16h* wq = (const v16h*)(A.DWQT + (size_t)tid * 128);
    float s = 0;
#pragma unroll
    for (int c = 0; c < 8; ++c) {
      v16h e = wq[c];
#pragma unroll
      for (int t2 = 0; t2 < 16; ++t2) s += sVec[c * 16 + t2] * (float)e[t2];
    }
    sVec[128 + tid] = s;
  }
  __syncthreads();
  for (int p = tid; p < 512; p += 256) {  // logits[hd][k]
    int hd = p >> 6, k = p & 63;
    float s = 0;
    for (int t2 = 0; t2 < 16; ++t2)
      s += sVec[128 + hd * 16 + t2] * sBq[k * 128 + hd * 16 + t2];
    sS[p] = (sTok[k] > 0.5f) ? s * 0.25f : NEGBIG;
  }
  __syncthreads();
  if (tid < 8) {  // softmax per head
    float mx = -3.4e38f;
    for (int k = 0; k < 64; ++k) mx = fmaxf(mx, sS[tid * 64 + k]);
    float su = 0.f;
    for (int k = 0; k < 64; ++k) {
      float e = __expf(sS[tid * 64 + k] - mx);
      sS[tid * 64 + k] = e;
      su += e;
    }
    float inv = 1.0f / su;
    for (int k = 0; k < 64; ++k) sS[tid * 64 + k] *= inv;
  }
  __syncthreads();
  if (tid < 128) {  // attn_out = P @ V_d  (pre-Wo; Wo applied batched in B)
    int hd = tid >> 4;
    float s = 0;
    for (int k = 0; k < 64; ++k) s += sS[hd * 64 + k] * sCk[k * 128 + tid];
    sVec[256 + tid] = s;
  }
  __syncthreads();
  // 12. export h_c, attn_out
  if (tid < 128) {
    A.hcws[(size_t)fb * 128 + tid] = sVec[tid];
    A.aows[(size_t)fb * 128 + tid] = sVec[256 + tid];
  }
  // 13. generator keys: EK = enc @ Wk_gen
  gemm_sw(sDv, 128, A.GWK, 8, 0, 0, sBq, 128, 64, 128, 128, nullptr, false);
  __syncthreads();
  for (int i = tid; i < 8192; i += 256)
    A.ekws[(size_t)fb * 8192 + i] = (half_t)sBq[i];
}

// ============== kernel B: batched decoder tail (64 sequences / WG) ==========
struct BArgs {
  const float *hcws, *aows;
  const half_t* ekws;
  const half_t *DWO, *DW1, *DW2, *GWQ;
  const float *db1, *db2, *dg1, *dbb1, *dg2, *dbb2;
  const int *nmask, *pmask;
  float* out;
};

__global__ __launch_bounds__(256) void dec_kernel(BArgs P) {
  __shared__ float sX[64 * 128];   // h_c -> FFN blk -> gq
  __shared__ float sAO[64 * 128];  // attn_out -> dec
  __shared__ float sT[64 * 128];   // t
  int tid = threadIdx.x;
  int w = tid >> 5;
  int fb0 = blockIdx.x * 64;

  for (int i = tid; i < 8192; i += 256) {
    sX[i] = P.hcws[(size_t)fb0 * 128 + i];
    sAO[i] = P.aows[(size_t)fb0 * 128 + i];
  }
  __syncthreads();
  // t = LN(h_c + attn_out @ Wo_d)
  gemm_sw(sAO, 128, P.DWO, 8, 0, 0, sT, 128, 64, 128, 128, nullptr, false);
  __syncthreads();
  ln_rows64(sX, sT, sT, P.dg1, P.dbb1);
  __syncthreads();
  // dec = LN(t + relu(t@W1+b1)@W2 + b2)
  v8f acc[4];
  {
    v8f z = {};
#pragma unroll
    for (int i = 0; i < 4; ++i) acc[i] = z;
  }
  for (int kb = 0; kb < 4; ++kb) {
    gemm_sw(sT, 128, P.DW1, 32, kb * 8, 0, sX, 128, 64, 128, 128,
            P.db1 + kb * 128, true);
    __syncthreads();
#pragma unroll
    for (int ti = 0; ti < 4; ++ti) {
      int t = w + ti * 8;
      int tm = t >> 3, tn = t & 7;
      for (int ks = 0; ks < 4; ++ks) {
        v16h a = frag_a_f32(sX, 128, tm * 16, ks * 32, 32);
        v16h b = frag_b_sw(P.DW2, 8, kb * 4 + ks, tn);
        acc[ti] = wmma16(a, b, acc[ti]);
      }
    }
    __syncthreads();
  }
#pragma unroll
  for (int ti = 0; ti < 4; ++ti) {
    int t = w + ti * 8;
    store_c(sAO, 128, (t >> 3) * 16, (t & 7) * 16, acc[ti], P.db2, false, 1.0f);
  }
  __syncthreads();
  ln_rows64(sT, sAO, sAO, P.dg2, P.dbb2);
  __syncthreads();
  // gq = dec @ Wq_gen
  gemm_sw(sAO, 128, P.GWQ, 8, 0, 0, sX, 128, 64, 128, 128, nullptr, false);
  __syncthreads();
  // generator logits + masks -> att_scores
  for (int p = tid; p < 4096; p += 256) {
    int r = p >> 6, k = p & 63;
    int fb = fb0 + r;
    int nag = fb >> 5, bb = fb & 31;
    bool ok = (P.pmask[bb * 64 + nag] != 0) &&
              (P.nmask[(bb * 64 + nag) * 64 + k] != 0);
    float s = NEGBIG;
    if (ok) {
      const v16h* ekv =
          (const v16h*)(P.ekws + (size_t)fb * 8192 + (size_t)k * 128);
      float a2 = 0;
#pragma unroll
      for (int c = 0; c < 8; ++c) {
        v16h e = ekv[c];
#pragma unroll
        for (int t2 = 0; t2 < 16; ++t2)
          a2 += sX[r * 128 + c * 16 + t2] * (float)e[t2];
      }
      s = a2 * 0.08838834764831845f;  // 1/sqrt(128)
    }
    P.out[(size_t)bb * 4096 + nag * 64 + k] = s;
  }
}

// ================== kernel C: masked average of h_c (no atomics) ============
__global__ void avg_kernel(const float* hcws, const int* pmask, float* out) {
  int b = blockIdx.x;
  int j = threadIdx.x;  // 128
  float np = 0.f, s = 0.f;
  for (int n = 0; n < 64; ++n) {
    int pm = pmask[b * 64 + n];
    np += (float)pm;
    if (pm != 0) s += hcws[(size_t)(n * 32 + b) * 128 + j];
  }
  out[131072 + b * 128 + j] = s / np;
}

// ================================= host =====================================
extern "C" void kernel_launch(void* const* d_in, const int* in_sizes, int n_in,
                              void* d_out, int out_size, void* d_ws,
                              size_t ws_size, hipStream_t stream) {
  (void)n_in; (void)out_size; (void)ws_size;
  // top-level input order (insertion vs sorted fallback via sizes)
  int iA = 0, iN = 1, iP = 2;
  if (in_sizes[1] == 32) { iN = 2; iP = 3; }  // sorted: agent,is_from,nmask,pmask
  const float* agent = (const float*)d_in[iA];
  const int* nmask = (const int*)d_in[iN];
  const int* pmask = (const int*)d_in[iP];
  const float* We = (const float*)d_in[4];
  const float* be = (const float*)d_in[5];
  const float* dW1 = (const float*)d_in[6];
  const float* dW2 = (const float*)d_in[7];
  const float* dWk = (const float*)d_in[8];
  const float* dWo = (const float*)d_in[9];
  const float* dWq = (const float*)d_in[10];
  const float* dWv = (const float*)d_in[11];
  const float* db1 = (const float*)d_in[12];
  const float* db2 = (const float*)d_in[13];
  const float* dbb1 = (const float*)d_in[14];
  const float* dbb2 = (const float*)d_in[15];
  const float* dg1 = (const float*)d_in[16];
  const float* dg2 = (const float*)d_in[17];
  const float* eW1 = (const float*)d_in[18];
  const float* eW2 = (const float*)d_in[19];
  const float* eWk = (const float*)d_in[20];
  const float* eWo = (const float*)d_in[21];
  const float* eWq = (const float*)d_in[22];
  const float* eWv = (const float*)d_in[23];
  const float* eb1 = (const float*)d_in[24];
  const float* eb2 = (const float*)d_in[25];
  const float* ebb1 = (const float*)d_in[26];
  const float* ebb2 = (const float*)d_in[27];
  const float* eg1 = (const float*)d_in[28];
  const float* eg2 = (const float*)d_in[29];
  const float* gWk = (const float*)d_in[30];
  const float* gWq = (const float*)d_in[31];

  // workspace layout (halves)
  half_t* w16 = (half_t*)d_ws;
  const half_t* WE = w16 + 0;
  const half_t* EWQ = w16 + 4096;
  const half_t* EWK = w16 + 20480;
  const half_t* EWV = w16 + 36864;
  const half_t* EWO = w16 + 53248;
  const half_t* EW1 = w16 + 69632;
  const half_t* EW2 = w16 + 135168;
  const half_t* DWK = w16 + 200704;
  const half_t* DWV = w16 + 217088;
  const half_t* DWQT = w16 + 233472;
  const half_t* DWO = w16 + 249856;
  const half_t* DW1 = w16 + 266240;
  const half_t* DW2 = w16 + 331776;
  const half_t* GWK = w16 + 397312;
  const half_t* GWQ = w16 + 413696;
  float* hcws = (float*)((char*)d_ws + 860160);
  float* aows = (float*)((char*)d_ws + 1908736);
  half_t* ekws = (half_t*)((char*)d_ws + 2957312);

  PrepArgs pp;
  pp.src[0] = We;  pp.src[1] = eWq; pp.src[2] = eWk; pp.src[3] = eWv;
  pp.src[4] = eWo; pp.src[5] = eW1; pp.src[6] = eW2; pp.src[7] = dWk;
  pp.src[8] = dWv; pp.src[9] = dWq; pp.src[10] = dWo; pp.src[11] = dW1;
  pp.src[12] = dW2; pp.src[13] = gWk; pp.src[14] = gWq;
  prep_kernel<<<256, 256, 0, stream>>>(pp, w16);

  AArgs a;
  a.agent = agent; a.nmask = nmask;
  a.WE = WE; a.be = be;
  a.EWQ = EWQ; a.EWK = EWK; a.EWV = EWV; a.EWO = EWO; a.EW1 = EW1; a.EW2 = EW2;
  a.eb1 = eb1; a.eb2 = eb2; a.eg1 = eg1; a.ebb1 = ebb1; a.eg2 = eg2;
  a.ebb2 = ebb2;
  a.DWK = DWK; a.DWV = DWV; a.DWQT = DWQT; a.GWK = GWK;
  a.hcws = hcws; a.aows = aows; a.ekws = ekws;
  enc_kernel<<<2048, 256, 0, stream>>>(a);

  BArgs bp;
  bp.hcws = hcws; bp.aows = aows; bp.ekws = ekws;
  bp.DWO = DWO; bp.DW1 = DW1; bp.DW2 = DW2; bp.GWQ = GWQ;
  bp.db1 = db1; bp.db2 = db2; bp.dg1 = dg1; bp.dbb1 = dbb1; bp.dg2 = dg2;
  bp.dbb2 = dbb2;
  bp.nmask = nmask; bp.pmask = pmask;
  bp.out = (float*)d_out;
  dec_kernel<<<32, 256, 0, stream>>>(bp);

  avg_kernel<<<32, 128, 0, stream>>>(hcws, pmask, (float*)d_out);
}
